// GCNBranch_neg_change_34437047780016
// MI455X (gfx1250) — 8-bit matrix operand order repaired and checked by reading, compiled, not run on hardware
//
#include <hip/hip_runtime.h>

#define NNODE 2048

typedef __attribute__((ext_vector_type(8))) int   v8i;
typedef __attribute__((ext_vector_type(8))) float v8f;
typedef __attribute__((ext_vector_type(2))) float v2f;

__global__ void binarize_kernel(const float* __restrict__ A_neg,
                                const float* __restrict__ A_pos,
                                unsigned char* __restrict__ ApT,
                                unsigned char* __restrict__ last,
                                unsigned char* __restrict__ uni) {
  int idx = blockIdx.x * blockDim.x + threadIdx.x;
  if (idx >= NNODE * NNODE) return;
  int i = idx >> 11;
  int j = idx & (NNODE - 1);
  unsigned char ap = A_pos[idx] > 0.0f ? 1 : 0;
  unsigned char an = A_neg[idx] > 0.0f ? 1 : 0;
  ApT[j * NNODE + i] = ap;
  last[idx] = an;
  uni[idx]  = an;
}

__global__ void deginv_kernel(const unsigned char* __restrict__ uni,
                              float* __restrict__ dinv) {
  int row = blockIdx.x * blockDim.x + threadIdx.x;
  if (row >= NNODE) return;
  const uint4* p = (const uint4*)(uni + (size_t)row * NNODE);
  unsigned s = 1;
  for (int i = 0; i < NNODE / 16; ++i) {
    uint4 q = p[i];
    s += __popc(q.x) + __popc(q.y) + __popc(q.z) + __popc(q.w);
  }
  dinv[row] = rsqrtf((float)s);
}

__global__ void expand_wmma_kernel(const unsigned char* __restrict__ last,
                                   const unsigned char* __restrict__ ApT,
                                   unsigned char* __restrict__ last_out,
                                   unsigned char* __restrict__ uni) {
  const int lane = threadIdx.x & 31;
  const int half = lane >> 4;
  const int l16  = lane & 15;
  const int tileM = blockIdx.x * 32;
  const int tileN = blockIdx.y * 64;

  const unsigned char* arow[2];
  const unsigned char* brow[4];
#pragma unroll
  for (int m = 0; m < 2; ++m)
    arow[m] = last + (size_t)(tileM + m * 16 + l16) * NNODE;
#pragma unroll
  for (int n = 0; n < 4; ++n)
    brow[n] = ApT + (size_t)(tileN + n * 16 + l16) * NNODE;

  v8i acc[2][4];
#pragma unroll
  for (int m = 0; m < 2; ++m)
#pragma unroll
    for (int n = 0; n < 4; ++n) acc[m][n] = {};

#pragma unroll 1
  for (int kb = 0; kb < NNODE; kb += 64) {
    v8i a[2], b[4];
#pragma unroll
    for (int i = 0; i < 8; ++i) {
      int ka = kb + ((i >> 1) << 4) + (half << 3) + ((i & 1) << 2);
      int kq = kb + ((i >> 1) << 4) + (half << 3) + ((i & 1) << 2);
#pragma unroll
      for (int m = 0; m < 2; ++m) a[m][i] = *(const int*)(arow[m] + ka);
#pragma unroll
      for (int n = 0; n < 4; ++n) b[n][i] = *(const int*)(brow[n] + kq);
    }
#pragma unroll
    for (int m = 0; m < 2; ++m)
#pragma unroll
      for (int n = 0; n < 4; ++n)
        acc[m][n] = __builtin_amdgcn_wmma_i32_16x16x64_iu8(
            false, a[m], false, b[n], acc[m][n], false, false);
  }

#pragma unroll
  for (int m = 0; m < 2; ++m)
#pragma unroll
    for (int n = 0; n < 4; ++n)
#pragma unroll
      for (int r = 0; r < 8; ++r) {
        size_t off = (size_t)(tileM + m * 16 + r + (half << 3)) * NNODE +
                     tileN + n * 16 + l16;
        unsigned char nv = acc[m][n][r] > 0 ? 1 : 0;
        last_out[off] = nv;
        uni[off] = uni[off] | nv;
      }
}

__global__ void agg_wmma_kernel(const unsigned char* __restrict__ uni,
                                const float* __restrict__ ts,
                                float* __restrict__ agg, int ldn) {
  const int lane = threadIdx.x & 31;
  const int half = lane >> 4;
  const int l16  = lane & 15;
  const int tileM = blockIdx.x * 32;
  const int tileN = blockIdx.y * 64;

  const unsigned char* arow[2];
#pragma unroll
  for (int m = 0; m < 2; ++m)
    arow[m] = uni + (size_t)(tileM + m * 16 + l16) * NNODE;

  v8f acc[2][4];
#pragma unroll
  for (int m = 0; m < 2; ++m)
#pragma unroll
    for (int n = 0; n < 4; ++n) acc[m][n] = {};

#pragma unroll 1
  for (int kb = 0; kb < NNODE; kb += 4) {
    v2f a[2], b[4];
#pragma unroll
    for (int i = 0; i < 2; ++i) {
      int k = kb + (half << 1) + i;
#pragma unroll
      for (int m = 0; m < 2; ++m) a[m][i] = (float)arow[m][k];
      const float* tr = ts + (size_t)k * ldn + tileN + l16;
#pragma unroll
      for (int n = 0; n < 4; ++n) b[n][i] = tr[n * 16];
    }
#pragma unroll
    for (int m = 0; m < 2; ++m)
#pragma unroll
      for (int n = 0; n < 4; ++n)
        acc[m][n] = __builtin_amdgcn_wmma_f32_16x16x4_f32(
            false, a[m], false, b[n], (short)0, acc[m][n], false, false);
  }
#pragma unroll
  for (int m = 0; m < 2; ++m)
#pragma unroll
    for (int n = 0; n < 4; ++n)
#pragma unroll
      for (int r = 0; r < 8; ++r)
        agg[(size_t)(tileM + m * 16 + r + (half << 3)) * ldn +
            tileN + n * 16 + l16] = acc[m][n][r];
}

__global__ void linear_kernel(const float* __restrict__ X,
                              const float* __restrict__ W,
                              const float* __restrict__ bias,
                              float* __restrict__ out, int M, int K, int N) {
  int idx = blockIdx.x * blockDim.x + threadIdx.x;
  if (idx >= M * N) return;
  int row = idx / N, col = idx % N;
  float acc = bias ? bias[col] : 0.0f;
  const float* xr = X + (size_t)row * K;
  for (int k = 0; k < K; ++k) acc += xr[k] * W[(size_t)k * N + col];
  out[idx] = acc;
}

__global__ void scale_pad_kernel(const float* __restrict__ t,
                                 const float* __restrict__ dinv,
                                 float* __restrict__ ts, int N, int ldn) {
  int idx = blockIdx.x * blockDim.x + threadIdx.x;
  if (idx >= NNODE * ldn) return;
  int row = idx / ldn, col = idx % ldn;
  ts[idx] = (col < N) ? dinv[row] * t[(size_t)row * N + col] : 0.0f;
}

__global__ void combine_kernel(const float* __restrict__ h,
                               const float* __restrict__ agg,
                               const float* __restrict__ ts,
                               const float* __restrict__ dinv,
                               const float* __restrict__ bg,
                               float* __restrict__ xout,
                               int N, int ldn, float w, int do_relu) {
  int idx = blockIdx.x * blockDim.x + threadIdx.x;
  if (idx >= NNODE * N) return;
  int row = idx / N, col = idx % N;
  size_t po = (size_t)row * ldn + col;
  float g = dinv[row] * (agg[po] + ts[po]) + bg[col];
  if (do_relu) g = fmaxf(g, 0.0f);
  xout[idx] = h[idx] + w * g;
}

extern "C" void kernel_launch(void* const* d_in, const int* in_sizes, int n_in,
                              void* d_out, int out_size, void* d_ws, size_t ws_size,
                              hipStream_t stream) {
  (void)in_sizes; (void)n_in; (void)out_size; (void)ws_size;

  const float* x    = (const float*)d_in[0];
  const float* Aneg = (const float*)d_in[1];
  const float* Apos = (const float*)d_in[2];
  const float* W_l1 = (const float*)d_in[3];  const float* b_l1 = (const float*)d_in[4];
  const float* W_l2 = (const float*)d_in[5];  const float* b_l2 = (const float*)d_in[6];
  const float* W_l3 = (const float*)d_in[7];  const float* b_l3 = (const float*)d_in[8];
  const float *W_g[6], *b_g[6];
  for (int s = 0; s < 6; ++s) { W_g[s] = (const float*)d_in[9 + 2 * s];
                                b_g[s] = (const float*)d_in[10 + 2 * s]; }

  char* ws = (char*)d_ws;
  const size_t MB = 1024 * 1024;
  unsigned char* ApT   = (unsigned char*)(ws + 0 * MB);
  unsigned char* lastA = (unsigned char*)(ws + 4 * MB);
  unsigned char* lastB = (unsigned char*)(ws + 8 * MB);
  unsigned char* uni   = (unsigned char*)(ws + 12 * MB);
  float* dinv = (float*)(ws + 16 * MB);
  float* hbuf = (float*)(ws + 17 * MB);
  float* tbuf = (float*)(ws + 19 * MB);
  float* tsb  = (float*)(ws + 21 * MB);
  float* aggb = (float*)(ws + 23 * MB);
  float* xa   = (float*)(ws + 25 * MB);
  float* xb   = (float*)(ws + 27 * MB);

  auto cdiv = [](int a, int b) { return (a + b - 1) / b; };

  binarize_kernel<<<cdiv(NNODE * NNODE, 256), 256, 0, stream>>>(Aneg, Apos, ApT,
                                                                lastA, uni);

  auto gcn_stage = [&](const float* gin, const float* Wg, const float* bg,
                       int N, float w, int relu, float* xout) {
    int ldn = (N + 63) & ~63;
    deginv_kernel<<<cdiv(NNODE, 256), 256, 0, stream>>>(uni, dinv);
    linear_kernel<<<cdiv(NNODE * N, 256), 256, 0, stream>>>(gin, Wg, nullptr,
                                                            tbuf, NNODE, N, N);
    scale_pad_kernel<<<cdiv(NNODE * ldn, 256), 256, 0, stream>>>(tbuf, dinv,
                                                                 tsb, N, ldn);
    agg_wmma_kernel<<<dim3(NNODE / 32, ldn / 64), 32, 0, stream>>>(uni, tsb,
                                                                   aggb, ldn);
    combine_kernel<<<cdiv(NNODE * N, 256), 256, 0, stream>>>(gin, aggb, tsb,
                                                             dinv, bg, xout,
                                                             N, ldn, w, relu);
  };
  auto expand = [&](const unsigned char* src, unsigned char* dst) {
    expand_wmma_kernel<<<dim3(NNODE / 32, NNODE / 64), 32, 0, stream>>>(src, ApT,
                                                                        dst, uni);
  };

  linear_kernel<<<cdiv(NNODE * 256, 256), 256, 0, stream>>>(x, W_l1, b_l1, hbuf,
                                                            NNODE, 512, 256);
  gcn_stage(hbuf, W_g[0], b_g[0], 256, 1.0f, 1, xa);

  expand(lastA, lastB);
  linear_kernel<<<cdiv(NNODE * 62, 256), 256, 0, stream>>>(xa, W_l2, b_l2, hbuf,
                                                           NNODE, 256, 62);
  gcn_stage(hbuf, W_g[1], b_g[1], 62, 1.0f, 1, xb);

  expand(lastB, lastA);
  linear_kernel<<<cdiv(NNODE * 64, 256), 256, 0, stream>>>(xb, W_l3, b_l3, hbuf,
                                                           NNODE, 62, 64);
  gcn_stage(hbuf, W_g[2], b_g[2], 64, 0.5f, 1, xa);

  expand(lastA, lastB);
  gcn_stage(xa, W_g[3], b_g[3], 64, 0.5f, 1, xb);

  expand(lastB, lastA);
  gcn_stage(xb, W_g[4], b_g[4], 64, 0.25f, 1, xa);

  expand(lastA, lastB);
  gcn_stage(xa, W_g[5], b_g[5], 64, 0.25f, 0, (float*)d_out);
}
